// LSTMLayer_21251498181293
// MI455X (gfx1250) — matrix operand order repaired and checked by reading, not compiled, not run on hardware
//
#include <hip/hip_runtime.h>

// LSTM forward for MI455X (gfx1250, wave32, WMMA).
// S=2048, B=64, D=512, H=512, gates 4H=2048.
//
// Strategy: latency-bound recurrence. 16 persistent workgroups (one per WGP),
// f16 weights resident in LDS (256KB/WGP), v_wmma_f32_16x16x32_f16 GEMM with
// f32 accumulators, cell state c in registers across all 2048 steps, h
// exchanged via a 64KB f16 fragment-layout double buffer in L2, one global
// barrier per step (monotonic atomic counter + acquire loads), and
// global_prefetch of x[s+1] overlapped with the step-s GEMM/barrier.

#define S_LEN 2048
#define BATCH 64
#define DIN   512
#define HID   512
#define NG    2048   // 4*H
#define NBLK  16
#define NTHR  512

typedef _Float16 v16h __attribute__((ext_vector_type(16)));
typedef float    v8f  __attribute__((ext_vector_type(8)));

// ---- workspace layout (bytes) ----
// pw    : f16 [NBLK][8 tiles][32 ksteps][32 lanes][16]   = 4 MB   (B fragments)
// hfrag : f16 [2][4 mt][16 ks][32 lanes][16]             = 128 KB (A fragments, double buffered)
// bias  : f32 [2048]                                     = 8 KB   (b_ih + b_hh)
// cnt   : u32                                                     (global barrier)
#define PW_HALVES_PER_BLK 131072              // 8*32*32*16
#define HFRAG_HALVES      32768               // 4*16*32*16
#define WS_PW_OFF   0
#define WS_HF_OFF   4194304
#define WS_BIAS_OFF (4194304 + 131072)
#define WS_CNT_OFF  (4194304 + 131072 + 8192)

__device__ __forceinline__ float sigm_f(float v) { return 1.0f / (1.0f + __expf(-v)); }
__device__ __forceinline__ float tanh_f(float v) { return 1.0f - 2.0f / (__expf(2.0f * v) + 1.0f); }

// ---------------- prep: bias sum, zero h buffer 0, zero barrier counter ----------------
__global__ void lstm_prep(const float* __restrict__ b_ih, const float* __restrict__ b_hh,
                          float* __restrict__ bias, unsigned long long* __restrict__ hz,
                          unsigned* __restrict__ cnt) {
    int g = blockIdx.x * blockDim.x + threadIdx.x;
    if (g < NG) bias[g] = b_ih[g] + b_hh[g];
    if (g < HFRAG_HALVES / 4) hz[g] = 0ull;   // zero hfrag buffer 0 (h_0 = 0)
    if (g == 0) *cnt = 0u;
}

// ---------------- pack weights f32 -> f16 in WMMA B-fragment order ----------------
// Combined K dim: k<512 -> W_ih[n][k], k>=512 -> W_hh[n][k-512].
// B fragment (16x16x32 f16): lane = nlocal + 16*((klocal/8)%2), idx = 8*(klocal/16) + klocal%8.
__global__ void lstm_pack(const float* __restrict__ Wih, const float* __restrict__ Whh,
                          _Float16* __restrict__ pw) {
    int g = blockIdx.x * blockDim.x + threadIdx.x;   // 0 .. 2M-1
    int k = g & 1023;
    int n = g >> 10;
    float w = (k < DIN) ? Wih[n * DIN + k] : Whh[n * HID + (k - DIN)];
    int gc = n >> 9, r = n & 511;
    int blk = r >> 5, j16 = (r >> 4) & 1, nl = n & 15;
    int t8 = gc * 2 + j16;
    int ks = k >> 5, kl = k & 31;
    int lane = nl + (((kl >> 3) & 1) << 4);
    int idx = ((kl >> 4) << 3) + (kl & 7);
    pw[(size_t)blk * PW_HALVES_PER_BLK + (size_t)((t8 * 32 + ks) * 32 + lane) * 16 + idx] =
        (_Float16)w;
}

// ---------------- main recurrence kernel: 16 blocks x 512 threads ----------------
__global__ __launch_bounds__(NTHR)
void lstm_rec(const float* __restrict__ x, const _Float16* __restrict__ pw,
              const float* __restrict__ bias, _Float16* __restrict__ hfrag,
              unsigned* __restrict__ cnt, float* __restrict__ out) {
    extern __shared__ char smem[];
    _Float16* lds_w = (_Float16*)smem;                 // 256 KB weight slice
    float*    lds_g = (float*)(smem + 262144);         // 32 KB gate exchange

    const int tid  = threadIdx.x;
    const int blk  = blockIdx.x;
    const int wave = tid >> 5;
    const int lane = tid & 31;
    const int gc = wave >> 2;   // which gate (i,f,g,o)
    const int mt = wave & 3;    // which 16-row M tile

    // Pull this block's 256KB weight slice into LDS once.
    {
        const uint4* src = (const uint4*)(pw + (size_t)blk * PW_HALVES_PER_BLK);
        uint4* dst = (uint4*)lds_w;
        for (int i = tid; i < PW_HALVES_PER_BLK * 2 / 16; i += NTHR) dst[i] = src[i];
    }
    __syncthreads();

    // elementwise mapping: thread -> (row em, 4 cols starting at ej0)
    const int em  = tid >> 3;            // 0..63 (batch row)
    const int ej0 = (tid & 7) << 2;      // 0..28 step 4 (local h col)
    const int emt = em >> 4, eml = em & 15;
    float creg[4] = {0.f, 0.f, 0.f, 0.f};

    const int mrow = mt * 16 + (lane & 15);  // A-matrix row (batch index)
    const int lh   = lane >> 4;              // lane-high bit for A fragment

    const int bw0 = ((gc * 2 + 0) * 32) * 32 * 16;   // LDS half-offset, B tile j16=0
    const int bw1 = ((gc * 2 + 1) * 32) * 32 * 16;   // j16=1

    for (int s = 0; s < S_LEN; ++s) {
        v8f acc0 = {};
        v8f acc1 = {};
        const _Float16* hbuf = hfrag + (size_t)(s & 1) * HFRAG_HALVES;
        const float* xrow = x + ((size_t)s * BATCH + mrow) * DIN;

        // Warm x[s+1] into WGP$/L2 while this step's GEMM + barrier run.
        // 512 threads x 2 cachelines (128B) cover the full 128KB x[s+1] slice.
        if (s + 1 < S_LEN) {
            const float* xn = x + (size_t)(s + 1) * BATCH * DIN + tid * 64;
            __builtin_prefetch(xn, 0, 1);
            __builtin_prefetch(xn + 32, 0, 1);
        }

        // ---- K part 1: x @ W_ih^T (ksteps 0..15), f32->f16 pack on the fly ----
#pragma unroll 4
        for (int ks = 0; ks < 16; ++ks) {
            const float* xp = xrow + ks * 32 + lh * 8;
            float4 c0 = *(const float4*)(xp);
            float4 c1 = *(const float4*)(xp + 4);
            float4 c2 = *(const float4*)(xp + 16);
            float4 c3 = *(const float4*)(xp + 20);
            v16h a;
            a[0]=(_Float16)c0.x;  a[1]=(_Float16)c0.y;  a[2]=(_Float16)c0.z;  a[3]=(_Float16)c0.w;
            a[4]=(_Float16)c1.x;  a[5]=(_Float16)c1.y;  a[6]=(_Float16)c1.z;  a[7]=(_Float16)c1.w;
            a[8]=(_Float16)c2.x;  a[9]=(_Float16)c2.y;  a[10]=(_Float16)c2.z; a[11]=(_Float16)c2.w;
            a[12]=(_Float16)c3.x; a[13]=(_Float16)c3.y; a[14]=(_Float16)c3.z; a[15]=(_Float16)c3.w;
            v16h b0 = *(const v16h*)(lds_w + bw0 + (ks * 32 + lane) * 16);
            v16h b1 = *(const v16h*)(lds_w + bw1 + (ks * 32 + lane) * 16);
            acc0 = __builtin_amdgcn_wmma_f32_16x16x32_f16(false, a, false, b0, (short)0, acc0, false, false);
            acc1 = __builtin_amdgcn_wmma_f32_16x16x32_f16(false, a, false, b1, (short)0, acc1, false, false);
        }
        // ---- K part 2: h @ W_hh^T (ksteps 16..31), A fragments precomputed f16 ----
#pragma unroll 4
        for (int ks = 16; ks < 32; ++ks) {
            v16h a = *(const v16h*)(hbuf + ((mt * 16 + (ks - 16)) * 32 + lane) * 16);
            v16h b0 = *(const v16h*)(lds_w + bw0 + (ks * 32 + lane) * 16);
            v16h b1 = *(const v16h*)(lds_w + bw1 + (ks * 32 + lane) * 16);
            acc0 = __builtin_amdgcn_wmma_f32_16x16x32_f16(false, a, false, b0, (short)0, acc0, false, false);
            acc1 = __builtin_amdgcn_wmma_f32_16x16x32_f16(false, a, false, b1, (short)0, acc1, false, false);
        }

        // ---- publish gate fragments to LDS ----
        *(v8f*)(lds_g + (size_t)((((gc * 4 + mt) * 2 + 0) * 32 + lane) * 8)) = acc0;
        *(v8f*)(lds_g + (size_t)((((gc * 4 + mt) * 2 + 1) * 32 + lane) * 8)) = acc1;
        __syncthreads();

        // ---- elementwise cell update (block-local: this block owns i,f,g,o of its 32 h cols) ----
        _Float16* hw = hfrag + (size_t)((s + 1) & 1) * HFRAG_HALVES;
#pragma unroll
        for (int u = 0; u < 4; ++u) {
            int j  = ej0 + u;          // local h col 0..31
            int jg = blk * 32 + j;     // global h col
            int j16 = j >> 4;
            int glane = (j & 15) + ((eml >> 3) << 4);
            int gv = eml & 7;
            int idx0 = ((emt * 2 + j16) * 32 + glane) * 8 + gv;
            float iv  = lds_g[idx0]        + bias[jg];
            float fv  = lds_g[idx0 + 2048] + bias[512 + jg];
            float gv_ = lds_g[idx0 + 4096] + bias[1024 + jg];
            float ov  = lds_g[idx0 + 6144] + bias[1536 + jg];
            iv = sigm_f(iv); fv = sigm_f(fv); ov = sigm_f(ov); gv_ = tanh_f(gv_);
            float cn = fv * creg[u] + iv * gv_;
            creg[u] = cn;
            float hn = ov * tanh_f(cn);
            out[(size_t)s * BATCH * HID + (size_t)em * HID + jg] = hn;
            if (s == S_LEN - 1) {
                out[(size_t)S_LEN * BATCH * HID + (size_t)em * HID + jg] = hn;                 // h_f
                out[(size_t)S_LEN * BATCH * HID + BATCH * HID + (size_t)em * HID + jg] = cn;   // c_f
            }
            // write h_new as f16 A-fragment: k = 512+jg -> kstep2 = blk, klocal = j
            int lane_h = eml + (((j >> 3) & 1) << 4);
            int idx_h  = ((j >> 4) << 3) + (j & 7);
            hw[((emt * 16 + blk) * 32 + lane_h) * 16 + idx_h] = (_Float16)hn;
        }

        // ---- device-wide barrier (monotonic counter, acquire spin) ----
        __threadfence();
        __syncthreads();
        if (tid == 0) {
            atomicAdd(cnt, 1u);
            unsigned target = (unsigned)(NBLK * (s + 1));
            while (__hip_atomic_load(cnt, __ATOMIC_ACQUIRE, __HIP_MEMORY_SCOPE_AGENT) < target)
                __builtin_amdgcn_s_sleep(1);
        }
        __syncthreads();
        __threadfence();
    }
}

extern "C" void kernel_launch(void* const* d_in, const int* in_sizes, int n_in,
                              void* d_out, int out_size, void* d_ws, size_t ws_size,
                              hipStream_t stream) {
    const float* x   = (const float*)d_in[0];
    const float* Wih = (const float*)d_in[1];
    const float* Whh = (const float*)d_in[2];
    const float* bih = (const float*)d_in[3];
    const float* bhh = (const float*)d_in[4];
    float* out = (float*)d_out;

    char* ws = (char*)d_ws;
    _Float16* pw    = (_Float16*)(ws + WS_PW_OFF);
    _Float16* hfrag = (_Float16*)(ws + WS_HF_OFF);
    float*    bias  = (float*)(ws + WS_BIAS_OFF);
    unsigned* cnt   = (unsigned*)(ws + WS_CNT_OFF);

    lstm_prep<<<32, 256, 0, stream>>>(bih, bhh, bias, (unsigned long long*)hfrag, cnt);
    lstm_pack<<<8192, 256, 0, stream>>>(Wih, Whh, pw);
    lstm_rec<<<NBLK, NTHR, 294912 /* 256KB weights + 32KB gates */, stream>>>(
        x, pw, bias, hfrag, cnt, out);
}
